// TimeAttentionBlock_19885698581047
// MI455X (gfx1250) — matrix operand order repaired and checked by reading, not compiled, not run on hardware
//
#include <hip/hip_runtime.h>
#include <hip/hip_bf16.h>
#include <stdint.h>

typedef __attribute__((ext_vector_type(16))) _Float16 v16h;
typedef __attribute__((ext_vector_type(8)))  _Float16 v8h;
typedef __attribute__((ext_vector_type(8)))  float    v8f;

#define T_LEN 1024
#define NHEADS 8
#define FD 64
#define NIN 64
#define BATCH 4

union AFrag { v16h v; v8h h[2]; };

__device__ __forceinline__ v8f wmma_f16f32(v16h a, v16h b, v8f c) {
  // D = A(16x32 f16) * B(32x16 f16) + C(16x16 f32)
  return __builtin_amdgcn_wmma_f32_16x16x32_f16(false, a, false, b, (short)0, c, false, false);
}

// A-fragment (16x32, f16) from row-major src: A[M][K] = base[(row0+M)*stride + k0 + K]
__device__ __forceinline__ v16h load_a_frag(const _Float16* base, int row0, int k0,
                                            int stride, int lane) {
  const int r = lane & 15, half = lane >> 4;
  const _Float16* p = base + (size_t)(row0 + r) * stride + (k0 + half * 16);
  AFrag f;
  f.h[0] = *(const v8h*)(p);
  f.h[1] = *(const v8h*)(p + 8);
  return f.v;
}

// B-fragment (32x16, f16) where B = src^T: B[K][col] = base[(col0+col)*stride + k0 + K]
__device__ __forceinline__ v16h load_bT_frag(const _Float16* base, int col0, int k0,
                                             int stride, int lane) {
  const int r = lane & 15, half = lane >> 4;
  const _Float16* p = base + (size_t)(col0 + r) * stride + (k0 + half * 16);
  return *(const v16h*)(p);
}

// ---- CDNA5 async global->LDS copy (ASYNCcnt-tracked DMA, per-lane 16B) ----
__device__ __forceinline__ void async_b128(uint32_t lds_byte_off, const void* g) {
  asm volatile("global_load_async_to_lds_b128 %0, %1, off"
               :: "v"(lds_byte_off), "v"((unsigned long long)(uintptr_t)g)
               : "memory");
}
__device__ __forceinline__ void wait_async_le4() {
#if __has_builtin(__builtin_amdgcn_s_wait_asynccnt)
  __builtin_amdgcn_s_wait_asynccnt(4);
#else
  asm volatile("s_wait_asynccnt 0x4" ::: "memory");
#endif
}
__device__ __forceinline__ void wait_async_le0() {
#if __has_builtin(__builtin_amdgcn_s_wait_asynccnt)
  __builtin_amdgcn_s_wait_asynccnt(0);
#else
  asm volatile("s_wait_asynccnt 0x0" ::: "memory");
#endif
}
__device__ __forceinline__ uint32_t lds_off_of(const void* p) {
  return (uint32_t)(uintptr_t)p;   // LDS aperture keeps offset in addr[31:0]
}

// ---------------- K0a: f32 -> f16 weight conversion ----------------
__global__ void k0_cvt(const float* __restrict__ src, _Float16* __restrict__ dst, int n) {
  int i = blockIdx.x * blockDim.x + threadIdx.x;
  if (i < n) dst[i] = (_Float16)src[i];
}

// ---------------- K0b: Toeplitz relative-position bias LUT ----------------
__global__ void k0_bias(const float* __restrict__ rel_bias, float* __restrict__ btab) {
  int i = blockIdx.x * blockDim.x + threadIdx.x;   // 8 * 2048
  int h = i >> 11, rp = i & 2047;
  if (rp >= 2047) return;
  int rel = rp - 1023;
  int ret = (rel >= 0) ? 16 : 0;
  int n = rel < 0 ? -rel : rel;
  int bucket;
  if (n < 8) {
    bucket = ret + n;
  } else {
    int vl = 8 + (int)(__logf((float)n * 0.125f) / __logf(16.0f) * 8.0f);
    if (vl > 15) vl = 15;
    bucket = ret + vl;
  }
  btab[h * 2048 + rp] = rel_bias[bucket * 8 + h];
}

// ---------------- K1: input head projection ----------------
__global__ void k1_inproj(const float* __restrict__ x, const float* __restrict__ w_in,
                          _Float16* __restrict__ xr_ct, _Float16* __restrict__ xr_tc) {
  int idx = blockIdx.x * blockDim.x + threadIdx.x;   // B*F*T
  int t = idx & (T_LEN - 1);
  int f = (idx >> 10) & (FD - 1);
  int b = idx >> 16;
  const float* xp = x + ((size_t)b * NIN * FD) * T_LEN + (size_t)f * T_LEN + t;
  float acc[NHEADS];
#pragma unroll
  for (int h = 0; h < NHEADS; ++h) acc[h] = 0.0f;
  for (int c = 0; c < NIN; ++c) {
    float xv = xp[(size_t)c * FD * T_LEN];
#pragma unroll
    for (int h = 0; h < NHEADS; ++h) acc[h] = fmaf(w_in[h * NIN + c], xv, acc[h]);
  }
#pragma unroll
  for (int h = 0; h < NHEADS; ++h) {
    _Float16 v = (_Float16)acc[h];
    xr_ct[((size_t)b * 512 + h * FD + f) * T_LEN + t] = v;
    xr_tc[((size_t)b * T_LEN + t) * 512 + h * FD + f] = v;
  }
}

// ---------------- K2: qk = w_qk @ xr (per b), M=1024, K=512, N=1024 ----------------
__global__ void k2_qk(const _Float16* __restrict__ wqk, const _Float16* __restrict__ xr_tc,
                      _Float16* __restrict__ qbuf, _Float16* __restrict__ kbuf) {
  int lane = threadIdx.x & 31;
  int wave = blockIdx.x * (blockDim.x >> 5) + (threadIdx.x >> 5);
  int ttile = wave & 63;
  int dtile = (wave >> 6) & 63;
  int b = wave >> 12;
  int m0 = dtile * 16, t0 = ttile * 16;
  const _Float16* xb = xr_tc + (size_t)b * T_LEN * 512;
  v8f acc = {};
  for (int k0 = 0; k0 < 512; k0 += 32) {
    v16h a = load_a_frag(wqk, m0, k0, 512, lane);
    v16h bm = load_bT_frag(xb, t0, k0, 512, lane);
    acc = wmma_f16f32(a, bm, acc);
  }
  int h = m0 >> 7;
  int rem = m0 & 127;
  bool isk = rem >= 64;
  int jb = rem & 63;
  int r0 = (lane >> 4) * 8;
  int t = t0 + (lane & 15);
  _Float16* dst = (isk ? kbuf : qbuf) +
                  (((size_t)(b * NHEADS + h) * T_LEN + t) * FD + jb + r0);
  v8h out;
#pragma unroll
  for (int j = 0; j < 8; ++j) out[j] = (_Float16)acc[j];
  *(v8h*)dst = out;
}

// ---------------- K3: fused flash attention ----------------
// 4 waves/block = 64 query rows sharing one (b,h); K/V tiles double-buffered in LDS
// via async global->LDS DMA, overlapped with WMMA + online softmax.
#define KSTR 72   // padded LDS row stride (halves) for K tile (32 x 64)
#define VSTR 40   // padded LDS row stride (halves) for V^T tile (64 x 32)

__global__ void __launch_bounds__(128) k3_attn(
    const _Float16* __restrict__ qbuf, const _Float16* __restrict__ kbuf,
    const _Float16* __restrict__ xr_ct, const float* __restrict__ btab,
    float* __restrict__ Ot) {
  __shared__ _Float16 kls[2][32 * KSTR];
  __shared__ _Float16 vls[2][64 * VSTR];
  __shared__ _Float16 pls[4][16 * 40];

  const int tid = threadIdx.x;
  const int lane = tid & 31;
  const int wv = tid >> 5;
  const int ngroup = blockIdx.x & 15;
  const int h = (blockIdx.x >> 4) & 7;
  const int b = blockIdx.x >> 7;
  const int n0 = ngroup * 64 + wv * 16;

  const _Float16* qb = qbuf + (size_t)(b * NHEADS + h) * T_LEN * FD;
  const _Float16* kb = kbuf + (size_t)(b * NHEADS + h) * T_LEN * FD;
  const _Float16* vb = xr_ct + ((size_t)b * 512 + h * FD) * T_LEN;  // V^T: [f][t]
  const float* bt = btab + h * 2048;

  const v16h qf0 = load_a_frag(qb, n0, 0, FD, lane);
  const v16h qf1 = load_a_frag(qb, n0, 32, FD, lane);

  const int r0 = (lane >> 4) * 8;
  const int cl = lane & 15;
  const float scale = 0.125f;   // 64^-0.5

  // per-thread async staging: K tile 32x64 (8 segs/row), V^T tile 64x32 (4 segs/row)
  const int ka0 = tid, ka1 = tid + 128;          // K transfer ids 0..255
  const int kr0 = ka0 >> 3, ks0 = ka0 & 7;
  const int kr1 = ka1 >> 3, ks1 = ka1 & 7;
  const int vr0 = ka0 >> 2, vs0 = ka0 & 3;
  const int vr1 = ka1 >> 2, vs1 = ka1 & 3;

  auto stage = [&](int buf, int m0) {
    async_b128(lds_off_of(&kls[buf][kr0 * KSTR + ks0 * 8]), kb + (size_t)(m0 + kr0) * FD + ks0 * 8);
    async_b128(lds_off_of(&kls[buf][kr1 * KSTR + ks1 * 8]), kb + (size_t)(m0 + kr1) * FD + ks1 * 8);
    async_b128(lds_off_of(&vls[buf][vr0 * VSTR + vs0 * 8]), vb + (size_t)vr0 * T_LEN + m0 + vs0 * 8);
    async_b128(lds_off_of(&vls[buf][vr1 * VSTR + vs1 * 8]), vb + (size_t)vr1 * T_LEN + m0 + vs1 * 8);
  };

  float M[8], L[8], alpha[8];
  v8f O[4] = {};
#pragma unroll
  for (int j = 0; j < 8; ++j) { M[j] = -3.0e38f; L[j] = 0.0f; }

  stage(0, 0);

  for (int it = 0; it < T_LEN / 32; ++it) {
    const int m0 = it * 32;
    const int cur = it & 1;
    if (it + 1 < T_LEN / 32) {
      stage(cur ^ 1, m0 + 32);   // overlap next-tile DMA with this tile's math
      wait_async_le4();          // oldest 4 (current buffer) are complete
    } else {
      wait_async_le0();
    }
    __syncthreads();             // current K/V tiles visible to all waves

    const _Float16* kt = kls[cur];
    const _Float16* vt = vls[cur];

    v8f S0 = {}, S1 = {};
    S0 = wmma_f16f32(qf0, load_bT_frag(kt, 0, 0, KSTR, lane), S0);
    S0 = wmma_f16f32(qf1, load_bT_frag(kt, 0, 32, KSTR, lane), S0);
    S1 = wmma_f16f32(qf0, load_bT_frag(kt, 16, 0, KSTR, lane), S1);
    S1 = wmma_f16f32(qf1, load_bT_frag(kt, 16, 32, KSTR, lane), S1);

    float s0[8], s1[8];
#pragma unroll
    for (int j = 0; j < 8; ++j) {
      int n = n0 + r0 + j;
      s0[j] = (S0[j] + bt[m0 + cl - n + 1023]) * scale;
      s1[j] = (S1[j] + bt[m0 + 16 + cl - n + 1023]) * scale;
    }
#pragma unroll
    for (int j = 0; j < 8; ++j) {
      float mx = fmaxf(s0[j], s1[j]);
      mx = fmaxf(mx, __shfl_xor(mx, 1, 32));
      mx = fmaxf(mx, __shfl_xor(mx, 2, 32));
      mx = fmaxf(mx, __shfl_xor(mx, 4, 32));
      mx = fmaxf(mx, __shfl_xor(mx, 8, 32));
      float nm = fmaxf(M[j], mx);
      float al = __expf(M[j] - nm);
      M[j] = nm;
      float p0 = __expf(s0[j] - nm);
      float p1 = __expf(s1[j] - nm);
      s0[j] = p0; s1[j] = p1;
      float sm = p0 + p1;
      sm += __shfl_xor(sm, 1, 32);
      sm += __shfl_xor(sm, 2, 32);
      sm += __shfl_xor(sm, 4, 32);
      sm += __shfl_xor(sm, 8, 32);
      L[j] = L[j] * al + sm;
      alpha[j] = al;
    }
    // P (16x32) -> per-wave LDS region for A-fragment re-layout
    _Float16* pw = pls[wv];
#pragma unroll
    for (int j = 0; j < 8; ++j) {
      pw[(r0 + j) * 40 + cl] = (_Float16)s0[j];
      pw[(r0 + j) * 40 + 16 + cl] = (_Float16)s1[j];
    }
    __syncthreads();
    AFrag pf;
    {
      const int r = lane & 15, half = lane >> 4;
      const _Float16* p = &pw[r * 40 + half * 16];
      pf.h[0] = *(const v8h*)(p);
      pf.h[1] = *(const v8h*)(p + 8);
    }
#pragma unroll
    for (int ft = 0; ft < 4; ++ft) {
      v8f o = O[ft];
#pragma unroll
      for (int j = 0; j < 8; ++j) o[j] *= alpha[j];
      v16h bv = load_bT_frag(vt, ft * 16, 0, VSTR, lane);
      O[ft] = wmma_f16f32(pf.v, bv, o);
    }
    __syncthreads();   // all waves done reading this buffer before it is re-staged
  }
  // epilogue: normalize, store transposed O_T[b][h][f][t]
  float inv[8];
#pragma unroll
  for (int j = 0; j < 8; ++j) inv[j] = 1.0f / L[j];
  float* ob = Ot + (size_t)(b * NHEADS + h) * FD * T_LEN;
#pragma unroll
  for (int ft = 0; ft < 4; ++ft) {
    int f = ft * 16 + cl;
    float* dp = ob + (size_t)f * T_LEN + n0 + r0;
    v8f o;
#pragma unroll
    for (int j = 0; j < 8; ++j) o[j] = O[ft][j] * inv[j];
    *(v8f*)dp = o;
  }
}

// ---------------- K4: output projection ----------------
__global__ void k4_outproj(const float* __restrict__ Ot, const float* __restrict__ w_out,
                           float* __restrict__ y) {
  int idx = blockIdx.x * blockDim.x + threadIdx.x;   // B*F*T
  int t = idx & (T_LEN - 1);
  int f = (idx >> 10) & (FD - 1);
  int b = idx >> 16;
  float o[NHEADS];
#pragma unroll
  for (int h = 0; h < NHEADS; ++h)
    o[h] = Ot[((size_t)(b * NHEADS + h) * FD + f) * T_LEN + t];
  for (int c = 0; c < NIN; ++c) {
    float acc = 0.0f;
#pragma unroll
    for (int h = 0; h < NHEADS; ++h) acc = fmaf(w_out[c * NHEADS + h], o[h], acc);
    y[((size_t)(b * NIN + c) * FD + f) * T_LEN + t] = acc;
  }
}

extern "C" void kernel_launch(void* const* d_in, const int* in_sizes, int n_in,
                              void* d_out, int out_size, void* d_ws, size_t ws_size,
                              hipStream_t stream) {
  (void)in_sizes; (void)n_in; (void)out_size; (void)ws_size;
  const float* x        = (const float*)d_in[0];
  const float* w_in     = (const float*)d_in[1];
  const float* w_qk     = (const float*)d_in[2];
  const float* w_out    = (const float*)d_in[3];
  const float* rel_bias = (const float*)d_in[4];
  float* y = (float*)d_out;

  char* ws = (char*)d_ws;
  size_t off = 0;
  auto wsalloc = [&](size_t bytes) -> void* {
    void* p = ws + off;
    off += (bytes + 255) & ~(size_t)255;
    return p;
  };
  _Float16* wqk16 = (_Float16*)wsalloc((size_t)1024 * 512 * 2);
  _Float16* xr_ct = (_Float16*)wsalloc((size_t)BATCH * 512 * T_LEN * 2);
  _Float16* xr_tc = (_Float16*)wsalloc((size_t)BATCH * T_LEN * 512 * 2);
  _Float16* qbuf  = (_Float16*)wsalloc((size_t)BATCH * NHEADS * T_LEN * FD * 2);
  _Float16* kbuf  = (_Float16*)wsalloc((size_t)BATCH * NHEADS * T_LEN * FD * 2);
  float*    btab  = (float*)wsalloc((size_t)NHEADS * 2048 * 4);
  float*    Ot    = (float*)wsalloc((size_t)BATCH * NHEADS * FD * T_LEN * 4);

  hipLaunchKernelGGL(k0_cvt,    dim3(2048), dim3(256), 0, stream, w_qk, wqk16, 1024 * 512);
  hipLaunchKernelGGL(k0_bias,   dim3(64),   dim3(256), 0, stream, rel_bias, btab);
  hipLaunchKernelGGL(k1_inproj, dim3(1024), dim3(256), 0, stream, x, w_in, xr_ct, xr_tc);
  hipLaunchKernelGGL(k2_qk,     dim3(2048), dim3(256), 0, stream, wqk16, xr_tc, qbuf, kbuf);
  hipLaunchKernelGGL(k3_attn,   dim3(512),  dim3(128), 0, stream, qbuf, kbuf, xr_ct, btab, Ot);
  hipLaunchKernelGGL(k4_outproj,dim3(1024), dim3(256), 0, stream, Ot, w_out, y);
}
